// AttentionGQA_59974923321580
// MI455X (gfx1250) — matrix operand order repaired and checked by reading, not compiled, not run on hardware
//
#include <hip/hip_runtime.h>
#include <hip/hip_bf16.h>

#define HIDDEN 2048
#define HEADS  32
#define KVH    8
#define HD     64
#define BATCH  2
#define SEQ    2048
#define MROWS  (BATCH*SEQ)

typedef __attribute__((ext_vector_type(16))) __bf16 v16bf;
typedef __attribute__((ext_vector_type(8)))  __bf16 v8bf;
typedef __attribute__((ext_vector_type(4)))  __bf16 v4bf;
typedef __attribute__((ext_vector_type(8)))  float  v8f;

static __device__ __forceinline__ v8f wmma_bf16(v16bf a, v16bf b, v8f c) {
  return __builtin_amdgcn_wmma_f32_16x16x32_bf16(false, a, false, b, (short)0, c, false, false);
}

static __device__ __forceinline__ v16bf pack16(v8bf lo, v8bf hi) {
  v16bf r;
#pragma unroll
  for (int e = 0; e < 8; ++e) { r[e] = lo[e]; r[e + 8] = hi[e]; }
  return r;
}

// CDNA5 async DMA: copy 16B from global to LDS, tracked by ASYNCcnt.
static __device__ __forceinline__ void async_ld_b128(unsigned ldsOff, const __bf16* g) {
  asm volatile("global_load_async_to_lds_b128 %0, %1, off" :: "v"(ldsOff), "v"(g) : "memory");
}
#define WAIT_ASYNC(n) asm volatile("s_wait_asynccnt " #n ::: "memory")

static __device__ __forceinline__ unsigned lds_off(const void* p) {
  return (unsigned)(unsigned long long)p;   // flat LDS addr: low 32 bits = LDS offset
}

// ---------------------------------------------------------------------------
// x (f32, row-major) -> bf16, streaming.
// ---------------------------------------------------------------------------
__global__ __launch_bounds__(256) void xconv_kernel(const float* __restrict__ x,
                                                    __bf16* __restrict__ xb) {
  const size_t i = ((size_t)blockIdx.x * 256 + threadIdx.x) * 8;
  const float4 a = *(const float4*)(x + i);
  const float4 b = *(const float4*)(x + i + 4);
  v8bf o;
  o[0]=(__bf16)a.x; o[1]=(__bf16)a.y; o[2]=(__bf16)a.z; o[3]=(__bf16)a.w;
  o[4]=(__bf16)b.x; o[5]=(__bf16)b.y; o[6]=(__bf16)b.z; o[7]=(__bf16)b.w;
  *(v8bf*)(xb + i) = o;
}

// ---------------------------------------------------------------------------
// W[K=2048][N] f32 -> WT[N][2048] bf16 (transpose via LDS tile, 32x32).
// ---------------------------------------------------------------------------
__global__ __launch_bounds__(256) void wconv_kernel(const float* __restrict__ W,
                                                    __bf16* __restrict__ WT, int N) {
  __shared__ __bf16 lt[32][36];
  const int k0 = blockIdx.x * 32;
  const int n0 = blockIdx.y * 32;
  const int r  = threadIdx.x >> 3;   // 0..31
  const int cg = threadIdx.x & 7;    // 0..7
  const float4 w4 = *(const float4*)(W + (size_t)(k0 + r) * N + n0 + cg * 4);
  lt[cg*4+0][r] = (__bf16)w4.x;  lt[cg*4+1][r] = (__bf16)w4.y;
  lt[cg*4+2][r] = (__bf16)w4.z;  lt[cg*4+3][r] = (__bf16)w4.w;
  __syncthreads();
  v4bf o;
#pragma unroll
  for (int j = 0; j < 4; ++j) o[j] = lt[r][cg*4 + j];
  *(v4bf*)(WT + (size_t)(n0 + r) * HIDDEN + k0 + cg * 4) = o;
}

// ---------------------------------------------------------------------------
// GEMM machinery: block tile 256x64, wave tile 32x64 (8 WMMA/step), K-step 32.
// bf16 A[M][2048] and WT[N][2048], contraction contiguous for both.
// Async double-buffered LDS; padded LDS rows (40 bf16) for bank spread.
// ---------------------------------------------------------------------------
#define GEMM_BODY(A_PTR, WT_PTR)                                               \
  __shared__ __bf16 As[2][256 * 40];                                           \
  __shared__ __bf16 Wsh[2][64 * 40];                                           \
  const int tid  = threadIdx.x;                                                \
  const int lane = tid & 31, wave = tid >> 5;                                  \
  const int n16  = lane & 15, half = lane >> 4;                                \
  const int mBase = blockIdx.x * 256, nBase = blockIdx.y * 64;                 \
  const int rC = tid >> 2, sC = tid & 3; /* 64 rows x 4 segs of 8 */           \
  const __bf16* aS = (A_PTR)  + (size_t)(mBase + rC) * HIDDEN + sC * 8;        \
  const __bf16* wS = (WT_PTR) + (size_t)(nBase + rC) * HIDDEN + sC * 8;        \
  unsigned aD[2], wD[2];                                                       \
  aD[0] = lds_off(&As[0][rC * 40 + sC * 8]);                                   \
  aD[1] = lds_off(&As[1][rC * 40 + sC * 8]);                                   \
  wD[0] = lds_off(&Wsh[0][rC * 40 + sC * 8]);                                  \
  wD[1] = lds_off(&Wsh[1][rC * 40 + sC * 8]);                                  \
  auto issue = [&](int k0, int bb) {                                           \
    async_ld_b128(aD[bb],                aS + k0);                             \
    async_ld_b128(aD[bb] + 64*40*2,      aS + (size_t)64  * HIDDEN + k0);      \
    async_ld_b128(aD[bb] + 128*40*2,     aS + (size_t)128 * HIDDEN + k0);      \
    async_ld_b128(aD[bb] + 192*40*2,     aS + (size_t)192 * HIDDEN + k0);      \
    async_ld_b128(wD[bb], wS + k0);                                            \
  };                                                                           \
  v8f acc[2][4] = {};                                                          \
  issue(0, 0);                                                                 \
  for (int k = 0; k < HIDDEN / 32; ++k) {                                      \
    const int bb = k & 1;                                                      \
    if (k < HIDDEN / 32 - 1) { issue((k + 1) * 32, bb ^ 1); WAIT_ASYNC(5); }   \
    else                     { WAIT_ASYNC(0); }                                \
    __syncthreads();                                                           \
    const __bf16* Ar0 = &As[bb][(wave * 32 + n16) * 40];                       \
    const __bf16* Ar1 = Ar0 + 16 * 40;                                         \
    const v16bf af0 = pack16(*(const v8bf*)(Ar0 + half * 8),                   \
                             *(const v8bf*)(Ar0 + 16 + half * 8));             \
    const v16bf af1 = pack16(*(const v8bf*)(Ar1 + half * 8),                   \
                             *(const v8bf*)(Ar1 + 16 + half * 8));             \
    v16bf bfr[4];                                                              \
    _Pragma("unroll")                                                          \
    for (int nt = 0; nt < 4; ++nt) {                                           \
      const __bf16* Wr = &Wsh[bb][(nt * 16 + n16) * 40 + half * 8];            \
      bfr[nt] = pack16(((const v8bf*)Wr)[0], ((const v8bf*)Wr)[2]);            \
    }                                                                          \
    _Pragma("unroll")                                                          \
    for (int nt = 0; nt < 4; ++nt) {                                           \
      acc[0][nt] = wmma_bf16(af0, bfr[nt], acc[0][nt]);                        \
      acc[1][nt] = wmma_bf16(af1, bfr[nt], acc[1][nt]);                        \
    }                                                                          \
    __syncthreads();                                                           \
  }

// ---------------------------------------------------------------------------
// Kernel: fused QKV GEMM.  N = 3072 concat of [Wq^T|Wk^T|Wv^T] rows.
// q -> [b,h,t,d], k -> [b,hk,t,d], v -> TRANSPOSED [b,hk,d,t] (all bf16).
// ---------------------------------------------------------------------------
__global__ __launch_bounds__(256) void qkv_gemm_kernel(
    const __bf16* __restrict__ xb, const __bf16* __restrict__ wt,
    __bf16* __restrict__ qo, __bf16* __restrict__ ko, __bf16* __restrict__ vo)
{
  GEMM_BODY(xb, wt)

  __bf16* outp; int col0, heads, vtrans;
  if (nBase < 2048)      { outp = qo; col0 = nBase;        heads = HEADS; vtrans = 0; }
  else if (nBase < 2560) { outp = ko; col0 = nBase - 2048; heads = KVH;   vtrans = 0; }
  else                   { outp = vo; col0 = nBase - 2560; heads = KVH;   vtrans = 1; }

#pragma unroll
  for (int su = 0; su < 2; ++su)
#pragma unroll
    for (int nt = 0; nt < 4; ++nt) {
      const int col = col0 + nt * 16 + n16;
      const int h = col >> 6, d = col & 63;
#pragma unroll
      for (int v = 0; v < 8; ++v) {
        const int r = mBase + wave * 32 + su * 16 + half * 8 + v;
        const int b = r >> 11, t = r & (SEQ - 1);
        const size_t idx = vtrans
            ? (((size_t)b * heads + h) * HD + d) * SEQ + t
            : (((size_t)b * heads + h) * SEQ + t) * HD + d;
        outp[idx] = (__bf16)acc[su][nt][v];
      }
    }
}

// ---------------------------------------------------------------------------
// Kernel: output projection GEMM.  out f32 [4096][2048] = y @ Wo.
// ---------------------------------------------------------------------------
__global__ __launch_bounds__(256) void out_gemm_kernel(
    const __bf16* __restrict__ yb, const __bf16* __restrict__ wto,
    float* __restrict__ out)
{
  GEMM_BODY(yb, wto)

#pragma unroll
  for (int su = 0; su < 2; ++su)
#pragma unroll
    for (int nt = 0; nt < 4; ++nt)
#pragma unroll
      for (int v = 0; v < 8; ++v) {
        const int r = mBase + wave * 32 + su * 16 + half * 8 + v;
        out[(size_t)r * HIDDEN + nBase + nt * 16 + n16] = acc[su][nt][v];
      }
}

// ---------------------------------------------------------------------------
// Kernel: causal flash attention, one wave = 16 query rows, 64 keys/step.
// Softmax tracked in the exp2 domain: one v_exp_f32 per probability.
// ---------------------------------------------------------------------------
__global__ __launch_bounds__(128) void attn_kernel(
    const __bf16* __restrict__ qb, const __bf16* __restrict__ kb,
    const __bf16* __restrict__ vt, __bf16* __restrict__ yb)
{
  __shared__ __bf16 plds[4][16 * 64];

  const int lane = threadIdx.x & 31;
  const int wave = threadIdx.x >> 5;
  const int n16  = lane & 15;
  const int half = lane >> 4;

  const int wl  = blockIdx.x * 4 + wave;
  const int qbk = wl & 127;
  const int h   = (wl >> 7) & 31;
  const int b   = wl >> 12;
  const int hk  = h >> 2;
  const int m0  = qbk * 16;

  const __bf16* qbase = qb + ((size_t)b * HEADS + h)  * SEQ * HD;
  const __bf16* kbase = kb + ((size_t)b * KVH  + hk) * SEQ * HD;
  const __bf16* vtb   = vt + ((size_t)b * KVH  + hk) * HD * SEQ;

  v16bf qa0, qa1;
  {
    const v8bf* p = (const v8bf*)(qbase + (size_t)(m0 + n16) * HD + half * 8);
    qa0 = pack16(p[0], p[2]);
    qa1 = pack16(p[4], p[6]);
  }

  const float SC = 0.18033688011112042f;   // (1/8) * log2(e)
  float mrow[8], lrow[8];
  v8f o[4] = {};
#pragma unroll
  for (int v = 0; v < 8; ++v) { mrow[v] = -3.0e38f; lrow[v] = 0.0f; }

  __bf16* L = plds[wave];
  const int kbMax = (qbk >> 2) + 1;        // 64-key blocks covering keys <= m0+15

  for (int kb4 = 0; kb4 < kbMax; ++kb4) {
    const int keyBase = kb4 * 64;

    // ---- S: 4 key-quarters x 2 d-halves = 8 WMMA
    v8f s[4] = {};
#pragma unroll
    for (int g = 0; g < 4; ++g) {
      const __bf16* kr = kbase + (size_t)(keyBase + g * 16 + n16) * HD;
      const v8bf* p = (const v8bf*)(kr + half * 8);
      s[g] = wmma_bf16(qa0, pack16(p[0], p[2]), s[g]);
      s[g] = wmma_bf16(qa1, pack16(p[4], p[6]), s[g]);
    }

    // ---- scale (exp2 domain) + causal mask + online softmax
    float pf[4][8], alpha[8];
#pragma unroll
    for (int v = 0; v < 8; ++v) {
      const int r = m0 + half * 8 + v;
      float sv[4];
#pragma unroll
      for (int g = 0; g < 4; ++g) {
        sv[g] = s[g][v] * SC;
        if (keyBase + g * 16 + n16 > r) sv[g] = -3.0e38f;
      }
      float tmax = fmaxf(fmaxf(sv[0], sv[1]), fmaxf(sv[2], sv[3]));
#pragma unroll
      for (int off = 1; off < 16; off <<= 1)
        tmax = fmaxf(tmax, __shfl_xor(tmax, off, 16));
      const float mnew = fmaxf(mrow[v], tmax);
      alpha[v] = exp2f(mrow[v] - mnew);
      float rsum = 0.0f;
#pragma unroll
      for (int g = 0; g < 4; ++g) { pf[g][v] = exp2f(sv[g] - mnew); rsum += pf[g][v]; }
#pragma unroll
      for (int off = 1; off < 16; off <<= 1)
        rsum += __shfl_xor(rsum, off, 16);
      lrow[v] = lrow[v] * alpha[v] + rsum;
      mrow[v] = mnew;
    }
#pragma unroll
    for (int dt = 0; dt < 4; ++dt)
#pragma unroll
      for (int v = 0; v < 8; ++v) o[dt][v] *= alpha[v];

    // ---- P: C-layout -> A-layout through LDS (intra-wave 16x64 transpose)
#pragma unroll
    for (int v = 0; v < 8; ++v) {
      const int r = half * 8 + v;
#pragma unroll
      for (int g = 0; g < 4; ++g)
        L[r * 64 + g * 16 + n16] = (__bf16)pf[g][v];
    }
    asm volatile("s_wait_dscnt 0" ::: "memory");
    v16bf pa0, pa1;
    {
      const __bf16* Pr = L + n16 * 64;
      pa0 = pack16(*(const v8bf*)(Pr + half * 8),      *(const v8bf*)(Pr + 16 + half * 8));
      pa1 = pack16(*(const v8bf*)(Pr + 32 + half * 8), *(const v8bf*)(Pr + 48 + half * 8));
    }

    // ---- O += P @ V  (V^T: contiguous key runs), 8 WMMA
#pragma unroll
    for (int dt = 0; dt < 4; ++dt) {
      const __bf16* vp = vtb + (size_t)(dt * 16 + n16) * SEQ + keyBase;
      const v8bf* q0 = (const v8bf*)(vp + half * 8);
      const v8bf* q1 = (const v8bf*)(vp + 32 + half * 8);
      o[dt] = wmma_bf16(pa0, pack16(q0[0], q0[2]), o[dt]);
      o[dt] = wmma_bf16(pa1, pack16(q1[0], q1[2]), o[dt]);
    }
  }

#pragma unroll
  for (int v = 0; v < 8; ++v) {
    const float inv = 1.0f / lrow[v];
    const int t = m0 + half * 8 + v;
    const size_t base = ((size_t)b * SEQ + t) * HIDDEN + h * HD;
#pragma unroll
    for (int dt = 0; dt < 4; ++dt)
      yb[base + dt * 16 + n16] = (__bf16)(o[dt][v] * inv);
  }
}

// ---------------------------------------------------------------------------
extern "C" void kernel_launch(void* const* d_in, const int* in_sizes, int n_in,
                              void* d_out, int out_size, void* d_ws, size_t ws_size,
                              hipStream_t stream) {
  const float* x  = (const float*)d_in[0];
  const float* Wq = (const float*)d_in[2];
  const float* Wk = (const float*)d_in[3];
  const float* Wv = (const float*)d_in[4];
  const float* Wo = (const float*)d_in[5];
  float* out = (float*)d_out;

  __bf16* ws = (__bf16*)d_ws;
  __bf16* xbuf = ws;                                    // 4096*2048   (aliased by ybuf)
  __bf16* wt   = xbuf + (size_t)MROWS * HIDDEN;         // 3072*2048   (aliased by wto)
  __bf16* qbuf = wt   + (size_t)3072 * HIDDEN;          // 2*32*2048*64
  __bf16* kbuf = qbuf + (size_t)BATCH * HEADS * SEQ * HD; // 2*8*2048*64
  __bf16* vbuf = kbuf + (size_t)BATCH * KVH * SEQ * HD;   // 2*8*64*2048 (transposed)
  __bf16* ybuf = xbuf;                                  // alias: x dead after QKV GEMM
  __bf16* wto  = wt;                                    // alias: QKV W dead after QKV GEMM

  // 1) convert x to bf16
  xconv_kernel<<<(MROWS * HIDDEN) / 8 / 256, 256, 0, stream>>>(x, xbuf);
  // 2) transpose+convert weights into concatenated [3072][2048] bf16
  wconv_kernel<<<dim3(64, 64), 256, 0, stream>>>(Wq, wt, 2048);
  wconv_kernel<<<dim3(64, 16), 256, 0, stream>>>(Wk, wt + (size_t)2048 * HIDDEN, 512);
  wconv_kernel<<<dim3(64, 16), 256, 0, stream>>>(Wv, wt + (size_t)2560 * HIDDEN, 512);
  // 3) fused QKV GEMM (async double-buffered LDS, WMMA)
  qkv_gemm_kernel<<<dim3(16, 48), 256, 0, stream>>>(xbuf, wt, qbuf, kbuf, vbuf);
  // 4) Wo transpose (reuses wt region; QKV weights dead now)
  wconv_kernel<<<dim3(64, 64), 256, 0, stream>>>(Wo, wto, 2048);
  // 5) flash attention (y aliases x region)
  attn_kernel<<<2048, 128, 0, stream>>>(qbuf, kbuf, vbuf, ybuf);
  // 6) output projection
  out_gemm_kernel<<<dim3(16, 32), 256, 0, stream>>>(ybuf, wto, out);
}
